// Mamba_45500883534552
// MI455X (gfx1250) — matrix operand order repaired and checked by reading, not compiled, not run on hardware
//
#include <hip/hip_runtime.h>
#include <stdint.h>

typedef __attribute__((ext_vector_type(16))) __bf16 v16bf;
typedef __attribute__((ext_vector_type(8)))  float  v8f;
typedef __attribute__((ext_vector_type(16))) unsigned short ush16;
typedef __attribute__((ext_vector_type(8)))  unsigned short ush8;

static __device__ __forceinline__ unsigned short f2bf(float f) {
  union { float f; unsigned int u; } v; v.f = f;
  unsigned int u = v.u;
  if ((u & 0x7F800000u) == 0x7F800000u) return (unsigned short)(u >> 16); // inf/nan
  return (unsigned short)((u + 0x7FFFu + ((u >> 16) & 1u)) >> 16);        // RNE
}

// Async copy 16B from global to LDS (per lane), tracked by ASYNCcnt.
static __device__ __forceinline__ void async_b128(unsigned int lds_addr,
                                                  const void* gaddr) {
  asm volatile("global_load_async_to_lds_b128 %0, %1, off"
               :: "v"(lds_addr), "v"((unsigned long long)(uintptr_t)gaddr)
               : "memory");
}
static __device__ __forceinline__ void wait_async0() {
  asm volatile("s_wait_asynccnt 0x0" ::: "memory");
}

// ---------------------------------------------------------------- conversions
__global__ void k_f32_to_bf16(const float* __restrict__ in,
                              unsigned short* __restrict__ out, int n) {
  int i = blockIdx.x * blockDim.x + threadIdx.x;
  if (i < n) out[i] = f2bf(in[i]);
}

// ------------------------------------------------- WMMA GEMM: C = A @ W^T
// A: [M, K] bf16 row-major, W: [N, K] bf16 row-major, C: [M, N] f32.
// M % 64 == 0, K % 32 == 0, any N (B rows clamped, stores guarded).
// Block 256 = 8 waves computing a 64x64 C block; async double-buffered LDS.
__global__ __launch_bounds__(256)
void k_gemm_bf16(const unsigned short* __restrict__ A,
                 const unsigned short* __restrict__ W,
                 float* __restrict__ C, int M, int N, int K) {
  __shared__ __align__(64) unsigned short Asb[2][64 * 32];
  __shared__ __align__(64) unsigned short Bsb[2][64 * 32];

  const int tid  = threadIdx.x;
  const int wave = tid >> 5;
  const int lane = tid & 31;
  const int l  = lane & 15;     // row (A) / col (B,C) within tile
  const int hf = lane >> 4;     // half-wave selector
  const int wm = wave >> 1;     // 0..3 : 16-row strip
  const int wn = wave & 1;      // 0..1 : 32-col strip
  const int bm = blockIdx.y;    // 64-row block
  const int bn = blockIdx.x;    // 64-col block

  // staging assignment: one 16B chunk of A and one of B per lane
  const int sr = tid >> 2;      // 0..63 row within block tile
  const int sc = tid & 3;       // 0..3  16B chunk within 64B row (32 halves)
  const int ar = bm * 64 + sr;
  int br = bn * 64 + sr; if (br >= N) br = N - 1;   // clamp (garbage cols unused)
  const unsigned short* gA = A + (size_t)ar * K + sc * 8;
  const unsigned short* gB = W + (size_t)br * K + sc * 8;
  const unsigned int lA0 = (unsigned int)(uintptr_t)(&Asb[0][sr * 32 + sc * 8]);
  const unsigned int lB0 = (unsigned int)(uintptr_t)(&Bsb[0][sr * 32 + sc * 8]);
  const unsigned int bufStride =
      (unsigned int)((uintptr_t)(&Asb[1][0]) - (uintptr_t)(&Asb[0][0]));

  v8f acc0 = {}, acc1 = {};
  const int nsteps = K >> 5;

  // prologue: stage k-step 0 into buffer 0
  async_b128(lA0, gA);
  async_b128(lB0, gB);

  for (int s = 0; s < nsteps; ++s) {
    wait_async0();
    __syncthreads();                       // staged data visible to all waves
    if (s + 1 < nsteps) {                  // prefetch next step into other buf
      const int koff = (s + 1) << 5;
      const unsigned int bsel = ((s + 1) & 1) ? bufStride : 0u;
      async_b128(lA0 + bsel, gA + koff);
      async_b128(lB0 + bsel, gB + koff);
    }
    const unsigned short* As = Asb[s & 1];
    const unsigned short* Bs = Bsb[s & 1];

    // A 16x32 fragment (shared by both output tiles of this wave)
    union { struct { ush8 a; ush8 b; } s2; v16bf v; } af;
    const unsigned short* arow = As + (wm * 16 + l) * 32;
    af.s2.a = *(const ush8*)(arow + hf * 16);
    af.s2.b = *(const ush8*)(arow + hf * 16 + 8);
    // two B 32x16 fragments
    union { ush16 u; v16bf v; } b0, b1;
    b0.u = *(const ush16*)(Bs + (wn * 32 + l) * 32 + hf * 16);
    b1.u = *(const ush16*)(Bs + (wn * 32 + 16 + l) * 32 + hf * 16);
    acc0 = __builtin_amdgcn_wmma_f32_16x16x32_bf16(false, af.v, false, b0.v,
                                                   (short)0, acc0, false, false);
    acc1 = __builtin_amdgcn_wmma_f32_16x16x32_bf16(false, af.v, false, b1.v,
                                                   (short)0, acc1, false, false);
    __syncthreads();                       // all reads done before next overwrite
  }

  // store: VGPR j -> row bm*64 + wm*16 + hf*8 + j, col bn*64 + wn*32 + c2*16 + l
  const int row0 = bm * 64 + wm * 16 + hf * 8;
  const int col0 = bn * 64 + wn * 32 + l;
#pragma unroll
  for (int j = 0; j < 8; ++j) {
    if (col0 < N)      C[(size_t)(row0 + j) * N + col0]      = acc0[j];
    if (col0 + 16 < N) C[(size_t)(row0 + j) * N + col0 + 16] = acc1[j];
  }
}

// -------------------------------------- causal depthwise conv (K=4) + SiLU
__global__ void k_conv_silu(const float* __restrict__ xz,
                            const float* __restrict__ cw,
                            const float* __restrict__ cb,
                            float* __restrict__ xs,
                            unsigned short* __restrict__ xs_bf,
                            int Bn, int L, int DI) {
  int i = blockIdx.x * blockDim.x + threadIdx.x;
  int total = Bn * L * DI;
  if (i >= total) return;
  int c = i % DI;
  int t = (i / DI) % L;
  int b = i / (DI * L);
  const float* w = cw + c * 4;
  float acc = cb[c];
#pragma unroll
  for (int j = 0; j < 4; ++j) {
    int tt = t - 3 + j;                           // left zero-pad by K-1
    if (tt >= 0) acc += w[j] * xz[(size_t)(b * L + tt) * 4096 + c];
  }
  float s = acc / (1.0f + __expf(-acc));          // silu
  xs[i] = s;
  xs_bf[i] = f2bf(s);
}

// ------------------------------------------ pack dt_low = x_dbl[:, 0:64] bf16
__global__ void k_extract_dtlow(const float* __restrict__ xdbl,
                                unsigned short* __restrict__ out, int rows) {
  int i = blockIdx.x * blockDim.x + threadIdx.x;
  if (i >= rows * 64) return;
  int r = i >> 6, c = i & 63;
  out[i] = f2bf(xdbl[(size_t)r * 96 + c]);
}

// -------------------------------------------------- dt = softplus(proj + b_dt)
__global__ void k_softplus_bias(float* __restrict__ dtp,
                                const float* __restrict__ b_dt,
                                int total, int DI) {
  int i = blockIdx.x * blockDim.x + threadIdx.x;
  if (i >= total) return;
  float v = dtp[i] + b_dt[i % DI];
  dtp[i] = (v > 20.0f) ? v : log1pf(__expf(v));
}

// ------------------------------------------------------- selective scan + gate
__global__ __launch_bounds__(256)
void k_scan(const float* __restrict__ xdbl,   // [B*L, 96] (B at 64, C at 80)
            const float* __restrict__ dt,     // [B*L, DI]
            const float* __restrict__ xs,     // [B*L, DI]
            const float* __restrict__ xz,     // [B*L, 4096], z at col 2048+
            const float* __restrict__ A_log,  // [DI, 16]
            const float* __restrict__ Dp,     // [DI]
            unsigned short* __restrict__ y_bf,// [B*L, DI] bf16
            int L, int DI) {
  const int c = blockIdx.x * blockDim.x + threadIdx.x;
  const int b = blockIdx.y;
  float Ac[16];
#pragma unroll
  for (int n = 0; n < 16; ++n) Ac[n] = -__expf(A_log[(size_t)c * 16 + n]);
  const float Dc = Dp[c];
  float h[16];
#pragma unroll
  for (int n = 0; n < 16; ++n) h[n] = 0.0f;

  __shared__ float sBC[32];
  for (int t = 0; t < L; ++t) {
    const size_t row = (size_t)(b * L + t);
    if (threadIdx.x < 32) sBC[threadIdx.x] = xdbl[row * 96 + 64 + threadIdx.x];
    __syncthreads();
    const float dtv = dt[row * DI + c];
    const float xv  = xs[row * DI + c];
    float y = Dc * xv;
#pragma unroll
    for (int n = 0; n < 16; ++n) {
      const float dA = __expf(dtv * Ac[n]);
      h[n] = dA * h[n] + (dtv * sBC[n]) * xv;
      y += h[n] * sBC[16 + n];
    }
    const float zv = xz[row * 4096 + 2048 + c];
    const float g  = zv / (1.0f + __expf(-zv));
    y_bf[row * DI + c] = f2bf(y * g);
    __syncthreads();
  }
}

// =============================================================== launch glue
extern "C" void kernel_launch(void* const* d_in, const int* in_sizes, int n_in,
                              void* d_out, int out_size, void* d_ws, size_t ws_size,
                              hipStream_t stream) {
  (void)in_sizes; (void)n_in; (void)out_size; (void)ws_size;
  const int Bn = 2, L = 1024, DM = 1024, DI = 2048;
  const int ML = Bn * L;  // 2048 rows

  const float* hs    = (const float*)d_in[0];
  const float* W_in  = (const float*)d_in[1];
  const float* convw = (const float*)d_in[2];
  const float* convb = (const float*)d_in[3];
  const float* W_x   = (const float*)d_in[4];
  const float* W_dt  = (const float*)d_in[5];
  const float* b_dt  = (const float*)d_in[6];
  const float* A_log = (const float*)d_in[7];
  const float* Dp    = (const float*)d_in[8];
  const float* W_out = (const float*)d_in[9];

  char* ws = (char*)d_ws;
  size_t off = 0;
  auto alloc = [&](size_t bytes) {
    char* p = ws + off;
    off = (off + bytes + 255) & ~(size_t)255;
    return p;
  };
  unsigned short* hs_bf    = (unsigned short*)alloc((size_t)ML * DM * 2);
  unsigned short* Win_bf   = (unsigned short*)alloc((size_t)2 * DI * DM * 2);
  unsigned short* Wx_bf    = (unsigned short*)alloc((size_t)96 * DI * 2);
  unsigned short* Wdt_bf   = (unsigned short*)alloc((size_t)DI * 64 * 2);
  unsigned short* Wout_bf  = (unsigned short*)alloc((size_t)DM * DI * 2);
  float*          xz       = (float*)alloc((size_t)ML * 2 * DI * 4);
  float*          xs       = (float*)alloc((size_t)ML * DI * 4);
  unsigned short* xs_bf    = (unsigned short*)alloc((size_t)ML * DI * 2);
  float*          xdbl     = (float*)alloc((size_t)ML * 96 * 4);
  unsigned short* dtlow_bf = (unsigned short*)alloc((size_t)ML * 64 * 2);
  float*          dtb      = (float*)alloc((size_t)ML * DI * 4);
  unsigned short* y_bf     = (unsigned short*)alloc((size_t)ML * DI * 2);

  auto cvt = [&](const float* s, unsigned short* d, int n) {
    k_f32_to_bf16<<<(n + 255) / 256, 256, 0, stream>>>(s, d, n);
  };
  cvt(hs, hs_bf, ML * DM);
  cvt(W_in, Win_bf, 2 * DI * DM);
  cvt(W_x, Wx_bf, 96 * DI);
  cvt(W_dt, Wdt_bf, DI * 64);
  cvt(W_out, Wout_bf, DM * DI);

  auto gemm = [&](const unsigned short* A, const unsigned short* W, float* C,
                  int M, int N, int K) {
    dim3 grid((N + 63) / 64, M / 64);
    k_gemm_bf16<<<grid, 256, 0, stream>>>(A, W, C, M, N, K);
  };

  // 1) xz = hs @ W_in^T                  [2048, 4096]
  gemm(hs_bf, Win_bf, xz, ML, 2 * DI, DM);
  // 2) x = silu(conv(x) + b)             [2048, 2048] f32 + bf16
  k_conv_silu<<<(Bn * L * DI + 255) / 256, 256, 0, stream>>>(
      xz, convw, convb, xs, xs_bf, Bn, L, DI);
  // 3) x_dbl = x @ W_x^T                 [2048, 96]
  gemm(xs_bf, Wx_bf, xdbl, ML, 96, DI);
  // 4) dt_low bf16 pack                  [2048, 64]
  k_extract_dtlow<<<(ML * 64 + 255) / 256, 256, 0, stream>>>(xdbl, dtlow_bf, ML);
  // 5) dt = softplus(dt_low @ W_dt^T + b_dt)   [2048, 2048]
  gemm(dtlow_bf, Wdt_bf, dtb, ML, DI, 64);
  k_softplus_bias<<<(ML * DI + 255) / 256, 256, 0, stream>>>(dtb, b_dt, ML * DI, DI);
  // 6) selective scan + D skip + silu(z) gating -> y bf16
  dim3 sg(DI / 256, Bn);
  k_scan<<<sg, 256, 0, stream>>>(xdbl, dtb, xs, xz, A_log, Dp, y_bf, L, DI);
  // 7) out = y @ W_out^T                 [2048, 1024] f32
  gemm(y_bf, Wout_bf, (float*)d_out, ML, DM, DI);
}
